// FeedForwardNet_72335839199562
// MI455X (gfx1250) — compile-verified, matrix operand order checked by reading, not run on hardware
//
#include <hip/hip_runtime.h>
#include <hip/hip_bf16.h>
#include <math.h>

// ---------------------------------------------------------------------------
// NEAT feed-forward DAG, CDNA5 (gfx1250) implementation.
//
// Decomposition per 128-node block t (prefix length s = 512 + 128*t):
//   Z[B,128]  = O[:, 0:s] @ W[block, 0:s]^T          (bf16 WMMA, fp32 accum)
//   then a sequential 128-step strictly-lower sweep inside the block.
// Each workgroup owns 16 batch rows; a row's activation history is produced
// only by that workgroup => zero inter-workgroup dependencies => one launch.
// ---------------------------------------------------------------------------

typedef __bf16 bf16_t;
typedef __attribute__((ext_vector_type(8)))  bf16_t v8bf;
typedef __attribute__((ext_vector_type(16))) bf16_t v16bf;
typedef __attribute__((ext_vector_type(8)))  float  v8f;

#define BATCH    4096
#define NIN      512
#define NTOT     2048
#define NEVAL    1536
#define NOUT     256
#define KB       128      // nodes per block
#define NBLK     12       // NEVAL / KB
#define MT       16       // batch rows per workgroup
#define NTHREADS 256      // 8 wave32 => 8 column tiles of 16

// ------------------------- prep: W fp32 -> bf16 ----------------------------
__global__ __launch_bounds__(256) void neat_prep_wbf16(const float* __restrict__ W,
                                                       bf16_t* __restrict__ Wbf,
                                                       int n) {
    int i = blockIdx.x * blockDim.x + threadIdx.x;
    int stride = gridDim.x * blockDim.x;
    for (; i < n; i += stride) Wbf[i] = (bf16_t)W[i];
}

// ------------------------------ main kernel --------------------------------
__global__ __launch_bounds__(NTHREADS) void neat_forward(
        const float* __restrict__ x,
        const float* __restrict__ bvec,
        float* __restrict__ out,
        bf16_t* __restrict__ O,          // [BATCH][NTOT] bf16 activation history
        const bf16_t* __restrict__ Wbf)  // [NEVAL][NTOT] bf16 weights
{
    __shared__ bf16_t sW[KB][KB];        // in-block strictly-lower weights (32 KB)
    __shared__ float  sZ[MT][KB + 4];    // fp32 pre-activations (~8.5 KB)
    __shared__ float  sO[MT];            // broadcast activation per sweep step

    const int tid  = threadIdx.x;
    const int wave = tid >> 5;           // 0..7 : column tile (16 nodes each)
    const int lane = tid & 31;
    const int hi   = lane >> 4;          // K-half select per ISA 16-bit layout
    const int lrow = lane & 15;          // matrix row (A: batch row, B: node)
    const int row0 = blockIdx.x * MT;

    // Stage 0: seed O with this workgroup's input rows (fp32 -> bf16).
    for (int i = tid; i < MT * NIN; i += NTHREADS) {
        int r = i >> 9;                  // / NIN
        int c = i & (NIN - 1);
        O[(size_t)(row0 + r) * NTOT + c] = (bf16_t)x[(size_t)(row0 + r) * NIN + c];
    }
    __threadfence_block();
    __syncthreads();

    for (int t = 0; t < NBLK; ++t) {
        const int s        = NIN + t * KB;   // dense-prefix length (mult. of 32)
        const int nodeBase = NIN + t * KB;   // first global node id of block

        // In-block weight tile -> LDS. sW[r][c] = W[node t*KB+r][col nodeBase+c]
        for (int i = tid; i < KB * KB; i += NTHREADS) {
            int r = i >> 7;
            int c = i & (KB - 1);
            sW[r][c] = Wbf[(size_t)(t * KB + r) * NTOT + (nodeBase + c)];
        }

        // Dense prefix GEMM: acc(16x16) += O[rows,k:k+32] * W[nodes,k:k+32]^T.
        // 16-bit operand striping (ISA 7.12.2): lane<16 holds K {0..7,16..23},
        // lane>=16 holds K {8..15,24..31}; both operands are row-gathers.
        v8f acc = {};
        const bf16_t* arow = O   + (size_t)(row0 + lrow) * NTOT;
        const bf16_t* brow = Wbf + (size_t)(t * KB + wave * 16 + lrow) * NTOT;
        for (int k = 0; k < s; k += 32) {
            union { v16bf v; v8bf h[2]; } a, b;
            a.h[0] = *(const v8bf*)(arow + k + hi * 8);        // global_load_b128
            a.h[1] = *(const v8bf*)(arow + k + 16 + hi * 8);
            b.h[0] = *(const v8bf*)(brow + k + hi * 8);
            b.h[1] = *(const v8bf*)(brow + k + 16 + hi * 8);
            acc = __builtin_amdgcn_wmma_f32_16x16x32_bf16(
                      /*neg_a=*/false, a.v, /*neg_b=*/false, b.v,
                      /*c_mod=*/(short)0, acc,
                      /*reuse_a=*/false, /*reuse_b=*/false);
        }

        // Spill C tiles to LDS (+bias). C layout: VGPR v -> row v + hi*8,
        // column = lane&15 within the wave's 16-wide tile.
        {
            const int   ccol = wave * 16 + lrow;
            const float bias = bvec[t * KB + ccol];
            #pragma unroll
            for (int v = 0; v < 8; ++v)
                sZ[v + hi * 8][ccol] = acc[v] + bias;
        }
        __syncthreads();

        // Sequential in-block sweep (the irreducible dependency chain).
        for (int jj = 0; jj < KB; ++jj) {
            const int gcol = nodeBase + jj;
            if (tid < MT) {
                float z = sZ[tid][jj] * 5.0f;
                z = fminf(fmaxf(z, -60.0f), 60.0f);
                float o = 1.0f / (1.0f + __expf(-z));
                sO[tid] = o;
                O[(size_t)(row0 + tid) * NTOT + gcol] = (bf16_t)o;
                int ev = t * KB + jj;
                if (ev >= NEVAL - NOUT)
                    out[(size_t)(row0 + tid) * NOUT + (ev - (NEVAL - NOUT))] = o;
            }
            __syncthreads();
            {
                // rank-1 update of the remaining columns (strictly lower).
                const int m = tid >> 4;          // batch row 0..15
                const int q = tid & 15;          // column phase
                const float o = sO[m];
                for (int c = jj + 1 + q; c < KB; c += 16)
                    sZ[m][c] += o * (float)sW[c][jj];
            }
            __syncthreads();
        }
        __threadfence_block();   // our bf16 O stores feed our own next GEMM
        __syncthreads();
    }
}

// ------------------------------- launcher ----------------------------------
extern "C" void kernel_launch(void* const* d_in, const int* in_sizes, int n_in,
                              void* d_out, int out_size, void* d_ws, size_t ws_size,
                              hipStream_t stream) {
    (void)in_sizes; (void)n_in; (void)out_size; (void)ws_size;
    const float* x = (const float*)d_in[0];
    const float* W = (const float*)d_in[1];
    const float* b = (const float*)d_in[2];
    float* out = (float*)d_out;

    // workspace layout: [0,16MB) O bf16 history; [16MB,22MB) W bf16
    bf16_t* O   = (bf16_t*)d_ws;
    bf16_t* Wbf = (bf16_t*)((char*)d_ws + (size_t)BATCH * NTOT * sizeof(bf16_t));

    neat_prep_wbf16<<<1024, 256, 0, stream>>>(W, Wbf, NEVAL * NTOT);
    neat_forward<<<BATCH / MT, NTHREADS, 0, stream>>>(x, b, out, O, Wbf);
}
